// WormholeTessellationExpert_29222957481987
// MI455X (gfx1250) — matrix operand order repaired and checked by reading, not compiled, not run on hardware
//
#include <hip/hip_runtime.h>
#include <hip/hip_bf16.h>
#include <math.h>

#define T_TILES 16
#define KW 4
#define TD 128
#define CTX 640          // (1+K)*td
#define HID 256
#define SEQ 4096
#define DIM 2048
#define BATCH 2

#define COMB_STRIDE 648  // 640 + 8 halves pad (16B-aligned rows, bank-spread)
#define H_STRIDE    264  // 256 + 8
#define B1S_STRIDE  24   // 16 + 8
#define W2S_STRIDE  136  // 128 + 8 (reuses comb region for layer 2)

typedef __attribute__((ext_vector_type(16))) __bf16   v16bf;
typedef __attribute__((ext_vector_type(8)))  float    v8f;
typedef __attribute__((ext_vector_type(4)))  unsigned int u32x4;
typedef __attribute__((ext_vector_type(4)))  float    f32x4;

union Frag16 { u32x4 u[2]; v16bf v; };

__device__ __forceinline__ unsigned short f2bf(float f) {
  unsigned int u = __float_as_uint(f);
  u += 0x7FFFu + ((u >> 16) & 1u);       // round-to-nearest-even
  return (unsigned short)(u >> 16);
}
__device__ __forceinline__ float bf2f(unsigned short h) {
  return __uint_as_float(((unsigned int)h) << 16);
}

// Branch-free exact-GELU: erf via Abramowitz-Stegun 7.1.26 (|err|<=1.5e-7,
// far below bf16 storage rounding).  One v_rcp_f32 + one native exp, no
// divergence, no EXEC manipulation.
__device__ __forceinline__ float gelu_exact(float x) {
  const float z  = x * 0.70710678118654752f;
  const float az = fabsf(z);
  const float t  = __builtin_amdgcn_rcpf(1.0f + 0.3275911f * az);
  float p = 1.061405429f;
  p = -1.453152027f + t * p;
  p =  1.421413741f + t * p;
  p = -0.284496736f + t * p;
  p =  0.254829592f + t * p;
  p = t * p;
  const float e  = __expf(-az * az);
  const float ea = 1.0f - p * e;               // erf(|z|)
  const float er = copysignf(ea, z);           // erf(z)  (bit ops, branchless)
  return 0.5f * x * (1.0f + er);
}

// ---------------------------------------------------------------------------
// Kernel 1: pack weights into bf16, K-major (so B-fragments are contiguous).
// W1t[k][n] = W1[n][k]  (640 x 256),  W2t[k][n] = W2[n][k]  (256 x 128)
// ---------------------------------------------------------------------------
__global__ void wte_pack_kernel(const float* __restrict__ W1,
                                const float* __restrict__ W2,
                                unsigned short* __restrict__ w1t,
                                unsigned short* __restrict__ w2t) {
  int r = blockIdx.x;
  int n = threadIdx.x;
  if (r < CTX) {
    w1t[r * HID + n] = f2bf(W1[(size_t)n * CTX + r]);
  } else {
    int rr = r - CTX;
    if (n < TD) w2t[rr * TD + n] = f2bf(W2[(size_t)n * HID + rr]);
  }
}

// ---------------------------------------------------------------------------
// Kernel 2: LayerNorm per (b,s) row of 2048, write bf16 x_norm.
// ---------------------------------------------------------------------------
__global__ void wte_ln_kernel(const float* __restrict__ x,
                              const float* __restrict__ gamma,
                              const float* __restrict__ beta,
                              unsigned short* __restrict__ xnorm) {
  __shared__ float rs[256];
  __shared__ float rq[256];
  const int tid = threadIdx.x;
  const size_t base = (size_t)blockIdx.x * DIM;
  const int c0 = tid * 8;

  f32x4 a = *(const f32x4*)(x + base + c0);
  f32x4 b = *(const f32x4*)(x + base + c0 + 4);
  float s = a.x + a.y + a.z + a.w + b.x + b.y + b.z + b.w;
  float q = a.x*a.x + a.y*a.y + a.z*a.z + a.w*a.w
          + b.x*b.x + b.y*b.y + b.z*b.z + b.w*b.w;
  rs[tid] = s; rq[tid] = q;
  __syncthreads();
  for (int off = 128; off > 0; off >>= 1) {
    if (tid < off) { rs[tid] += rs[tid + off]; rq[tid] += rq[tid + off]; }
    __syncthreads();
  }
  const float mean = rs[0] * (1.0f / DIM);
  const float var  = rq[0] * (1.0f / DIM) - mean * mean;
  const float rstd = rsqrtf(var + 1e-5f);

  float vals[8] = {a.x, a.y, a.z, a.w, b.x, b.y, b.z, b.w};
  union { unsigned short h[8]; u32x4 u; } ov;
  #pragma unroll
  for (int i = 0; i < 8; i++) {
    float yv = (vals[i] - mean) * rstd * gamma[c0 + i] + beta[c0 + i];
    ov.h[i] = f2bf(yv);
  }
  *(u32x4*)(xnorm + base + c0) = ov.u;
}

// ---------------------------------------------------------------------------
// Kernel 3: tile_repr[b][t][d] = mean over S of x_norm tile.  32 blocks.
// ---------------------------------------------------------------------------
__global__ void wte_tilemean_kernel(const unsigned short* __restrict__ xnorm,
                                    float* __restrict__ tr) {
  __shared__ float red[256];
  const int bt = blockIdx.x;
  const int b = bt >> 4, t = bt & 15;
  const int tid = threadIdx.x;
  const int d = tid & 127, sg = tid >> 7;
  float acc = 0.0f;
  const size_t base = (size_t)b * SEQ * DIM + t * TD + d;
  for (int s = sg; s < SEQ; s += 2) acc += bf2f(xnorm[base + (size_t)s * DIM]);
  red[tid] = acc;
  __syncthreads();
  if (tid < 128) tr[bt * TD + tid] = (red[tid] + red[tid + 128]) * (1.0f / SEQ);
}

// ---------------------------------------------------------------------------
// Kernel 4: routing — q/k matvecs, l2norm, scores, top-4.  One block.
// ---------------------------------------------------------------------------
__global__ void wte_route_kernel(const float* __restrict__ tr,
                                 const float* __restrict__ Wq,
                                 const float* __restrict__ bq,
                                 const float* __restrict__ Wk,
                                 const float* __restrict__ bk,
                                 int* __restrict__ routes) {
  __shared__ float trL[BATCH * T_TILES * TD];
  __shared__ float qL [BATCH * T_TILES * TD];
  __shared__ float kL [BATCH * T_TILES * TD];
  __shared__ float scL[BATCH * T_TILES * T_TILES];
  const int tid = threadIdx.x;

  for (int i = tid; i < BATCH * T_TILES * TD; i += 256) trL[i] = tr[i];
  __syncthreads();

  for (int i = tid; i < BATCH * T_TILES * TD; i += 256) {
    const int d = i & 127, bt = i >> 7;
    const float* v  = &trL[bt * TD];
    const float* wq = &Wq[(size_t)d * TD];
    const float* wk = &Wk[(size_t)d * TD];
    float aq = bq[d], ak = bk[d];
    for (int j = 0; j < TD; j++) { float xv = v[j]; aq += xv * wq[j]; ak += xv * wk[j]; }
    qL[i] = aq; kL[i] = ak;
  }
  __syncthreads();

  if (tid < BATCH * T_TILES) {
    float sq = 0.0f, sk = 0.0f;
    for (int j = 0; j < TD; j++) {
      float a = qL[tid * TD + j]; sq += a * a;
      float c = kL[tid * TD + j]; sk += c * c;
    }
    const float inq = 1.0f / fmaxf(sqrtf(sq), 1e-12f);
    const float ink = 1.0f / fmaxf(sqrtf(sk), 1e-12f);
    for (int j = 0; j < TD; j++) { qL[tid * TD + j] *= inq; kL[tid * TD + j] *= ink; }
  }
  __syncthreads();

  for (int i = tid; i < BATCH * T_TILES * T_TILES; i += 256) {
    const int u = i & 15, t = (i >> 4) & 15, b = i >> 8;
    float s = 0.0f;
    for (int j = 0; j < TD; j++)
      s += qL[(b * T_TILES + t) * TD + j] * kL[(b * T_TILES + u) * TD + j];
    scL[i] = (u == t) ? -1e9f : s;
  }
  __syncthreads();

  if (tid < BATCH * T_TILES) {
    float sc[T_TILES];
    for (int u = 0; u < T_TILES; u++) sc[u] = scL[tid * T_TILES + u];
    for (int k = 0; k < KW; k++) {
      float best = -3.0e38f; int bi = 0;
      for (int u = 0; u < T_TILES; u++)
        if (sc[u] > best) { best = sc[u]; bi = u; }   // ties -> lowest index
      routes[tid * KW + k] = bi;
      sc[bi] = -3.0e38f;
    }
  }
}

// ---------------------------------------------------------------------------
// Kernel 5: main MLP with bf16 WMMA.  grid = (S/128, T, B), 256 thr = 8 waves.
// Each wave owns 16 s-rows (one M-tile).  LDS: combined rows + per-ntile W1t
// slab (layer 1); comb region is reused to hold the whole W2t for layer 2.
// ---------------------------------------------------------------------------
__global__ void __launch_bounds__(256, 1)
wte_mlp_kernel(const unsigned short* __restrict__ xnorm,
               const int* __restrict__ routes,
               const unsigned short* __restrict__ w1t,
               const unsigned short* __restrict__ w2t,
               const float* __restrict__ b1,
               const float* __restrict__ b2,
               const float* __restrict__ x,
               float* __restrict__ out) {
  extern __shared__ unsigned short smem[];
  unsigned short* comb = smem;                         // 128 x 648 (layer1 A)
  unsigned short* hbuf = smem + 128 * COMB_STRIDE;     // 128 x 264
  unsigned short* b1s  = hbuf + 128 * H_STRIDE;        // 640 x 24

  const int tid  = threadIdx.x;
  const int wave = tid >> 5;
  const int lane = tid & 31;
  const int hi   = lane >> 4;     // lane group (selects K sub-block in A frag)
  const int ln   = lane & 15;     // N / M-low index

  const int s0 = blockIdx.x * 128;
  const int t  = blockIdx.y;
  const int b  = blockIdx.z;

  int src[5];
  src[0] = t;
  #pragma unroll
  for (int k = 0; k < KW; k++) src[k + 1] = routes[(b * T_TILES + t) * KW + k];

  // ---- stage combined [128][640] = [own | r0 | r1 | r2 | r3] into LDS ----
  const size_t xbase = ((size_t)b * SEQ + s0) * DIM;
  #pragma unroll
  for (int j = 0; j < 5; j++) {
    const unsigned short* sp = xnorm + xbase + src[j] * TD;
    for (int idx = tid; idx < 128 * 16; idx += 256) {
      const int row = idx >> 4, c = idx & 15;
      u32x4 v = *(const u32x4*)(sp + (size_t)row * DIM + c * 8);
      *(u32x4*)(comb + row * COMB_STRIDE + j * TD + c * 8) = v;
    }
  }

  const int aRow = wave * 16 + ln;
  const unsigned short* ar = comb + aRow * COMB_STRIDE;

  // ---- layer 1: [128,640] @ W1t -> gelu -> hbuf [128,256] ----
  for (int nt = 0; nt < T_TILES; nt++) {
    __syncthreads();   // combined staged (1st iter) / b1s reads done (later)
    for (int r = tid; r < CTX; r += 256) {
      const unsigned short* sp = w1t + (size_t)r * HID + nt * 16;
      unsigned short* dp = b1s + r * B1S_STRIDE;
      *(u32x4*)(dp)     = *(const u32x4*)(sp);
      *(u32x4*)(dp + 8) = *(const u32x4*)(sp + 8);
    }
    __syncthreads();

    const float bias = b1[nt * 16 + ln];
    v8f acc = { bias, bias, bias, bias, bias, bias, bias, bias };
    #pragma unroll 4
    for (int kb = 0; kb < CTX / 32; kb++) {
      Frag16 af, bf;
      const int ko = kb * 32 + hi * 8;
      af.u[0] = *(const u32x4*)(ar + ko);
      af.u[1] = *(const u32x4*)(ar + ko + 16);
      const unsigned short* br = b1s + ko * B1S_STRIDE + ln;
      #pragma unroll
      for (int s = 0; s < 16; s += 2) {
        const int kk = (s & 7) + ((s >> 3) << 4);
        bf.u[s >> 3][(s >> 1) & 3] = (unsigned int)br[kk * B1S_STRIDE] | ((unsigned int)br[(kk + 1) * B1S_STRIDE] << 16);
      }
      acc = __builtin_amdgcn_wmma_f32_16x16x32_bf16(false, af.v, false, bf.v,
                                                    (short)0, acc, false, false);
    }
    #pragma unroll
    for (int i = 0; i < 8; i++) {             // branch-free GELU epilogue
      const float g = gelu_exact(acc[i]);
      const int row = wave * 16 + i + hi * 8;
      hbuf[row * H_STRIDE + nt * 16 + ln] = f2bf(g);
    }
  }

  // ---- stage W2t [256][128] into the (now dead) comb region ----
  __syncthreads();
  unsigned short* w2s = comb;                          // 256 x 136
  {
    const unsigned short* sp = w2t + (size_t)tid * TD;
    unsigned short* dp = w2s + tid * W2S_STRIDE;
    #pragma unroll
    for (int c = 0; c < 16; c++)
      *(u32x4*)(dp + c * 8) = *(const u32x4*)(sp + c * 8);
  }
  __syncthreads();

  // ---- layer 2: hbuf [128,256] @ W2t -> +bias -> residual add ----
  const unsigned short* hr = hbuf + aRow * H_STRIDE;
  for (int nt = 0; nt < TD / 16; nt++) {
    const float bias = b2[nt * 16 + ln];
    v8f acc = { bias, bias, bias, bias, bias, bias, bias, bias };
    #pragma unroll
    for (int kb = 0; kb < HID / 32; kb++) {
      Frag16 af, bf;
      const int ko = kb * 32 + hi * 8;
      af.u[0] = *(const u32x4*)(hr + ko);
      af.u[1] = *(const u32x4*)(hr + ko + 16);
      const unsigned short* br = w2s + ko * W2S_STRIDE + nt * 16 + ln;
      #pragma unroll
      for (int s = 0; s < 16; s += 2) {
        const int kk = (s & 7) + ((s >> 3) << 4);
        bf.u[s >> 3][(s >> 1) & 3] = (unsigned int)br[kk * W2S_STRIDE] | ((unsigned int)br[(kk + 1) * W2S_STRIDE] << 16);
      }
      acc = __builtin_amdgcn_wmma_f32_16x16x32_bf16(false, af.v, false, bf.v,
                                                    (short)0, acc, false, false);
    }
    #pragma unroll
    for (int i = 0; i < 8; i++) {
      const int row = wave * 16 + i + hi * 8;
      const size_t gi = ((size_t)b * SEQ + s0 + row) * DIM + t * TD + nt * 16 + ln;
      out[gi] = x[gi] + acc[i];
    }
  }
}

// ---------------------------------------------------------------------------
// launch
// ---------------------------------------------------------------------------
extern "C" void kernel_launch(void* const* d_in, const int* in_sizes, int n_in,
                              void* d_out, int out_size, void* d_ws, size_t ws_size,
                              hipStream_t stream) {
  const float* x     = (const float*)d_in[0];
  const float* gamma = (const float*)d_in[1];
  const float* beta  = (const float*)d_in[2];
  const float* Wq    = (const float*)d_in[3];
  const float* bq    = (const float*)d_in[4];
  const float* Wk    = (const float*)d_in[5];
  const float* bk    = (const float*)d_in[6];
  const float* W1    = (const float*)d_in[7];
  const float* b1    = (const float*)d_in[8];
  const float* W2    = (const float*)d_in[9];
  const float* b2    = (const float*)d_in[10];
  float* out = (float*)d_out;

  char* ws = (char*)d_ws;
  unsigned short* xnorm = (unsigned short*)(ws);                 // 33,554,432 B
  float*          tr    = (float*)(ws + 33554432);               //     16,384 B
  int*            routes= (int*)(ws + 33570816);                 //        512 B
  unsigned short* w1t   = (unsigned short*)(ws + 33571328);      //    327,680 B
  unsigned short* w2t   = (unsigned short*)(ws + 33899008);      //     65,536 B

  wte_pack_kernel<<<dim3(CTX + HID), dim3(256), 0, stream>>>(W1, W2, w1t, w2t);
  wte_ln_kernel<<<dim3(BATCH * SEQ), dim3(256), 0, stream>>>(x, gamma, beta, xnorm);
  wte_tilemean_kernel<<<dim3(BATCH * T_TILES), dim3(256), 0, stream>>>(xnorm, tr);
  wte_route_kernel<<<dim3(1), dim3(256), 0, stream>>>(tr, Wq, bq, Wk, bk, routes);

  const size_t lds_bytes =
      (size_t)(128 * COMB_STRIDE + 128 * H_STRIDE + CTX * B1S_STRIDE) * sizeof(unsigned short); // 264,192 B
  wte_mlp_kernel<<<dim3(SEQ / 128, T_TILES, BATCH), dim3(256), lds_bytes, stream>>>(
      xnorm, routes, w1t, w2t, b1, b2, x, out);
}
